// PGE_60748017434901
// MI455X (gfx1250) — compile-verified, matrix operand order checked by reading, not run on hardware
//
#include <hip/hip_runtime.h>
#include <hip/hip_bf16.h>
#include <stdint.h>

// Problem constants (from reference): x:(1024,128), W1:(128,256), W2:(128,128), W3:(1,128)
constexpr int N   = 1024;
constexpr int H   = 128;
constexpr int F   = 128;
constexpr int JT  = 64;         // j-rows per pass block
constexpr int NJB = N / JT;     // 16 j-blocks
constexpr int NBLK = N * NJB;   // 16384 pass blocks
constexpr int LDH = 136;        // padded LDS row stride (f16 elems): 272B = 68 banks -> conflict-free b128 frag loads

typedef _Float16 half8  __attribute__((ext_vector_type(8)));
typedef _Float16 half16 __attribute__((ext_vector_type(16)));
typedef float    float8 __attribute__((ext_vector_type(8)));

// 16-bit A/B fragment per CDNA5 ISA 7.12.2: lane (ln,g) holds 8 contiguous f16 at
// k = kk + g*8 (VGPR0-3) and 8 more at k = kk + 16 + g*8 (VGPR4-7).
__device__ __forceinline__ half16 ldfrag(const _Float16* __restrict__ row, int klo) {
  union { half16 v; half8 h[2]; } u;
  u.h[0] = *reinterpret_cast<const half8*>(row + klo);
  u.h[1] = *reinterpret_cast<const half8*>(row + klo + 16);
  return u.v;
}

// ---------------- Stage 0: A = x @ Wa^T, B = x @ Wb^T  (67 MFLOP, negligible) ----------------
__global__ __launch_bounds__(256) void pge_in_gemm(const float* __restrict__ x,
                                                   const float* __restrict__ W1,
                                                   float* __restrict__ Am,
                                                   float* __restrict__ Bm) {
  __shared__ float sx[F];
  const int row = blockIdx.x;
  const int t = threadIdx.x;
  if (t < F) sx[t] = x[row * F + t];
  __syncthreads();
  const int h = t & (H - 1);
  const int half_sel = t >> 7;                       // 0 -> Wa (cols 0..127), 1 -> Wb (cols 128..255)
  const float* __restrict__ w = W1 + h * (2 * F) + half_sel * F;
  float acc = 0.f;
#pragma unroll 16
  for (int f = 0; f < F; ++f) acc = fmaf(sx[f], w[f], acc);
  if (half_sel == 0) Am[row * H + h] = acc;
  else               Bm[row * H + h] = acc;
}

// ---------------- Stage 0b: BN1 stats factorized: var(A_j+B_i) = varA + varB ----------------
__global__ __launch_bounds__(128) void pge_stats1(const float* __restrict__ Am,
                                                  const float* __restrict__ Bm,
                                                  const float* __restrict__ g1,
                                                  const float* __restrict__ be1,
                                                  float* __restrict__ s1,
                                                  float* __restrict__ t1) {
  const int c = threadIdx.x;
  float sa = 0.f, qa = 0.f, sb = 0.f, qb = 0.f;
  for (int j = 0; j < N; ++j) {
    const float a = Am[j * H + c]; sa += a; qa = fmaf(a, a, qa);
    const float b = Bm[j * H + c]; sb += b; qb = fmaf(b, b, qb);
  }
  const float inv = 1.f / (float)N;
  const float ma = sa * inv, mb = sb * inv;
  const float va = qa * inv - ma * ma, vb = qb * inv - mb * mb;
  const float s = g1[c] * rsqrtf(va + vb + 1e-5f);   // b1 cancels inside BN
  s1[c] = s;
  t1[c] = be1[c] - (ma + mb) * s;
}

// ---------------- W2 -> f16 ----------------
__global__ __launch_bounds__(256) void pge_w2cvt(const float* __restrict__ W2,
                                                 _Float16* __restrict__ w2h) {
  const int e = blockIdx.x * 256 + threadIdx.x;
  w2h[e] = (_Float16)W2[e];
}

// ---------------- Main fused WMMA GEMM: y[i,j,k] = sum_c relu(s1*(A[j]+B[i])+t1)[c] * W2[k,c]
// Register-blocked: K outermost, 8 independent 16x16 accumulators per wave so the
// 8 WMMAs per K-step have no RAW chain and B-fragment ds_loads can be pipelined.
// PASS==1: accumulate per-channel sum / sumsq of y (BN2 stats, b2 cancels).
// PASS==2: fused BN2 + ReLU + dot with W3 -> logits[i,j].
template <int PASS>
__global__ __launch_bounds__(128) void pge_pass(const float* __restrict__ Am,
                                                const float* __restrict__ Bm,
                                                const float* __restrict__ s1,
                                                const float* __restrict__ t1,
                                                const _Float16* __restrict__ w2h,
                                                const float* __restrict__ s2,
                                                const float* __restrict__ t2,
                                                const float* __restrict__ W3,
                                                float* __restrict__ psum,
                                                float* __restrict__ psq,
                                                float* __restrict__ lgt) {
  __shared__ _Float16 sh1[JT * LDH];    // 17408 B : h1 tile (64 x 128 f16)
  __shared__ _Float16 sw2[H * LDH];     // 34816 B : W2 (128 x 128 f16)
  __shared__ float sBr[H], ss1[H], st1[H];
  __shared__ float saux[3 * H];         // s2 | t2 | w3 (pass 2)
  __shared__ float sWsum[4 * H], sWsq[4 * H];

  const int i  = blockIdx.x;            // 0..1023 (B row)
  const int j0 = blockIdx.y * JT;       // j tile base (A rows)
  const int t  = threadIdx.x;           // 128 threads = 4 wave32

  // Pull the tile working set toward the caches while we stage small arrays.
  __builtin_prefetch(&Am[(j0 + (t >> 1)) * H + (t & 1) * 64], 0, 1);
  __builtin_prefetch(&w2h[t * H], 0, 1);

  sBr[t] = Bm[i * H + t];
  ss1[t] = s1[t];
  st1[t] = t1[t];
  if constexpr (PASS == 2) {
    saux[t] = s2[t]; saux[H + t] = t2[t]; saux[2 * H + t] = W3[t];
  }
  __syncthreads();

  // Build h1 tile in LDS (BN1 + ReLU fused), float4 global reads.
  for (int e = t; e < JT * (H / 4); e += 128) {
    const int jj = e >> 5;
    const int c  = (e & 31) * 4;
    const float4 a4 = *reinterpret_cast<const float4*>(&Am[(j0 + jj) * H + c]);
    _Float16* d = &sh1[jj * LDH + c];
    d[0] = (_Float16)fmaxf(fmaf(ss1[c + 0], a4.x + sBr[c + 0], st1[c + 0]), 0.f);
    d[1] = (_Float16)fmaxf(fmaf(ss1[c + 1], a4.y + sBr[c + 1], st1[c + 1]), 0.f);
    d[2] = (_Float16)fmaxf(fmaf(ss1[c + 2], a4.z + sBr[c + 2], st1[c + 2]), 0.f);
    d[3] = (_Float16)fmaxf(fmaf(ss1[c + 3], a4.w + sBr[c + 3], st1[c + 3]), 0.f);
  }
  // Stage W2 tile (dword copies; L2-resident across all 16384 blocks).
  {
    const uint32_t* __restrict__ w2u = reinterpret_cast<const uint32_t*>(w2h);
    for (int e = t; e < H * (H / 2); e += 128) {
      const int k = e >> 6;
      const int c = (e & 63) * 2;
      *reinterpret_cast<uint32_t*>(&sw2[k * LDH + c]) = w2u[e];
    }
  }
  __syncthreads();

  const int wv   = t >> 5;              // wave -> 16-row strip
  const int lane = t & 31;
  const int ln   = lane & 15;
  const int g    = lane >> 4;
  const int m0   = wv * 16;
  const _Float16* __restrict__ arow = &sh1[(m0 + ln) * LDH];
  const _Float16* __restrict__ brow = &sw2[ln * LDH];   // +ns*16*LDH per strip

  float8 acc[8];
#pragma unroll
  for (int ns = 0; ns < 8; ++ns) acc[ns] = {};

#pragma unroll
  for (int kk = 0; kk < H; kk += 32) {
    const half16 af = ldfrag(arow, kk + g * 8);
#pragma unroll
    for (int ns = 0; ns < 8; ++ns) {
      const half16 bf = ldfrag(brow + ns * (16 * LDH), kk + g * 8);
      acc[ns] = __builtin_amdgcn_wmma_f32_16x16x32_f16(false, af, false, bf,
                                                       (short)0, acc[ns], false, false);
    }
  }

  // C/D layout: element v of lane (ln,g) is row m0 + v + g*8, col ns*16 + ln.
  if constexpr (PASS == 1) {
#pragma unroll
    for (int ns = 0; ns < 8; ++ns) {
      float ps = 0.f, pq = 0.f;
#pragma unroll
      for (int v = 0; v < 8; ++v) { const float y = acc[ns][v]; ps += y; pq = fmaf(y, y, pq); }
      ps += __shfl_xor(ps, 16);
      pq += __shfl_xor(pq, 16);
      if (g == 0) { sWsum[wv * H + ns * 16 + ln] = ps; sWsq[wv * H + ns * 16 + ln] = pq; }
    }
    __syncthreads();
    const float s = sWsum[t] + sWsum[H + t] + sWsum[2 * H + t] + sWsum[3 * H + t];
    const float q = sWsq[t]  + sWsq[H + t]  + sWsq[2 * H + t]  + sWsq[3 * H + t];
    const int b = i * NJB + blockIdx.y;
    psum[b * H + t] = s;
    psq [b * H + t] = q;
  } else {
    float lg0[8];
#pragma unroll
    for (int v = 0; v < 8; ++v) lg0[v] = 0.f;
#pragma unroll
    for (int ns = 0; ns < 8; ++ns) {
      const int n = ns * 16 + ln;
      const float sc = saux[n], tc = saux[H + n], w3c = saux[2 * H + n];
#pragma unroll
      for (int v = 0; v < 8; ++v) {
        const float h2 = fmaxf(fmaf(sc, acc[ns][v], tc), 0.f);
        lg0[v] = fmaf(h2, w3c, lg0[v]);
      }
    }
    // Reduce logit partials across the 16 lanes of each half-wave group.
#pragma unroll
    for (int m = 1; m < 16; m <<= 1) {
#pragma unroll
      for (int v = 0; v < 8; ++v) lg0[v] += __shfl_xor(lg0[v], m);
    }
    if (ln == 0) {
      const int jr = j0 + m0 + g * 8;
#pragma unroll
      for (int v = 0; v < 8; ++v) lgt[i * N + jr + v] = lg0[v];
    }
  }
}

// ---------------- BN2 coefficients from partials (deterministic tree reduce) ----------------
__global__ __launch_bounds__(1024) void pge_stats2(const float* __restrict__ psum,
                                                   const float* __restrict__ psq,
                                                   const float* __restrict__ g2,
                                                   const float* __restrict__ be2,
                                                   float* __restrict__ s2,
                                                   float* __restrict__ t2) {
  __shared__ float rs[8 * H], rq[8 * H];
  const int t = threadIdx.x;
  const int c = t & (H - 1);
  const int part = t >> 7;              // 8 partitions over the 16384 blocks
  const int bpp = NBLK / 8;
  float s = 0.f, q = 0.f;
  for (int b = part * bpp; b < (part + 1) * bpp; ++b) {
    s += psum[b * H + c];
    q += psq [b * H + c];
  }
  rs[part * H + c] = s; rq[part * H + c] = q;
  __syncthreads();
  if (t < H) {
    float S = 0.f, Q = 0.f;
#pragma unroll
    for (int p = 0; p < 8; ++p) { S += rs[p * H + t]; Q += rq[p * H + t]; }
    const float inv = 1.f / ((float)N * (float)N);
    const float mean = S * inv;
    const float var  = Q * inv - mean * mean;
    const float sc = g2[t] * rsqrtf(var + 1e-5f);    // b2 cancels inside BN
    s2[t] = sc;
    t2[t] = be2[t] - mean * sc;
  }
}

// ---------------- adj = sigmoid(0.5*(L + L^T) + b3), zero diagonal ----------------
__global__ __launch_bounds__(256) void pge_final(const float* __restrict__ lgt,
                                                 const float* __restrict__ b3,
                                                 float* __restrict__ out) {
  const int idx = blockIdx.x * 256 + threadIdx.x;
  const int i = idx >> 10, j = idx & (N - 1);
  const float L = 0.5f * (lgt[i * N + j] + lgt[j * N + i]) + b3[0];
  out[idx] = (i == j) ? 0.f : (1.f / (1.f + __expf(-L)));
}

extern "C" void kernel_launch(void* const* d_in, const int* in_sizes, int n_in,
                              void* d_out, int out_size, void* d_ws, size_t ws_size,
                              hipStream_t stream) {
  const float* x   = (const float*)d_in[0];
  const float* W1  = (const float*)d_in[1];
  // d_in[2] = b1 : cancels inside BN1
  const float* W2  = (const float*)d_in[3];
  // d_in[4] = b2 : cancels inside BN2
  const float* W3  = (const float*)d_in[5];
  const float* b3  = (const float*)d_in[6];
  const float* g1  = (const float*)d_in[7];
  const float* be1 = (const float*)d_in[8];
  const float* g2  = (const float*)d_in[9];
  const float* be2 = (const float*)d_in[10];
  float* out = (float*)d_out;

  // Workspace layout (bytes), ~22 MB total.
  char* ws = (char*)d_ws;
  float*    Am   = (float*)(ws + 0);          // 1024*128 f32
  float*    Bm   = (float*)(ws + 524288);     // 1024*128 f32
  float*    s1   = (float*)(ws + 1048576);
  float*    t1   = (float*)(ws + 1049088);
  float*    s2   = (float*)(ws + 1049600);
  float*    t2   = (float*)(ws + 1050112);
  _Float16* w2h  = (_Float16*)(ws + 1050624); // 128*128 f16
  float*    psum = (float*)(ws + 1083392);    // 16384*128 f32
  float*    psq  = (float*)(ws + 9472000);    // 16384*128 f32
  float*    lgt  = (float*)(ws + 17860608);   // 1024*1024 f32

  pge_in_gemm<<<N, 256, 0, stream>>>(x, W1, Am, Bm);
  pge_stats1<<<1, 128, 0, stream>>>(Am, Bm, g1, be1, s1, t1);
  pge_w2cvt<<<(H * H) / 256, 256, 0, stream>>>(W2, w2h);
  dim3 grid(N, NJB);
  pge_pass<1><<<grid, 128, 0, stream>>>(Am, Bm, s1, t1, w2h, s2, t2, W3, psum, psq, lgt);
  pge_stats2<<<1, 1024, 0, stream>>>(psum, psq, g2, be2, s2, t2);
  pge_pass<2><<<grid, 128, 0, stream>>>(Am, Bm, s1, t1, w2h, s2, t2, W3, psum, psq, lgt);
  pge_final<<<(N * N) / 256, 256, 0, stream>>>(lgt, b3, out);

  (void)in_sizes; (void)n_in; (void)out_size; (void)ws_size;
}
